// MultiHeadSpatialAttention_88691074662880
// MI455X (gfx1250) — compile-verified, matrix operand order checked by reading, not run on hardware
//
#include <hip/hip_runtime.h>
#include <hip/hip_bf16.h>

// ---------------------------------------------------------------------------
// MultiHeadSpatialAttention on MI455X (gfx1250): bf16 WMMA everywhere,
// double-buffered LDS with async global->LDS staging when available.
// ---------------------------------------------------------------------------

typedef __bf16 v16bf __attribute__((ext_vector_type(16)));
typedef float  v8f   __attribute__((ext_vector_type(8)));
typedef int    v4i   __attribute__((vector_size(4 * sizeof(int))));

#define EMBED 1024
#define HEADS 16
#define HDIM  64
#define BATCH 256
#define SEQ   100
#define M_TOT (BATCH * SEQ)   // 25600 rows
#define SP    112             // seq padded to 7 x 16 tiles
#define KP    128             // k-dim padded for P@V (4 x 32)

#define LDA   40              // LDS row stride (bf16 elems), pads banks
#define TILE_A (128 * LDA)    // one A tile in LDS (bf16 elems)
#define TILE_B (64 * LDA)     // one B tile in LDS

#if __has_builtin(__builtin_amdgcn_global_load_async_to_lds_b128) && \
    __has_builtin(__builtin_amdgcn_s_wait_asynccnt)
#define GEMM_ASYNC 1
#else
#define GEMM_ASYNC 0
#endif

#define AS1 __attribute__((address_space(1)))
#define AS3 __attribute__((address_space(3)))

typedef AS1 v4i* gptr4;   // global int4*  (clang prints AS1 as __device__)
typedef AS3 v4i* lptr4;   // LDS int4*

// Generic->AS1 / AS3 casts (LDS generic address: low 32 bits == LDS offset).
__device__ __forceinline__ gptr4 gcast(const void* p) {
    return (gptr4)(unsigned long long)p;
}
__device__ __forceinline__ lptr4 lcast(void* p) {
    return (lptr4)(unsigned int)(unsigned long long)p;
}

__device__ __forceinline__ unsigned short f32_to_bf16(float f) {
    unsigned int u = __float_as_uint(f);
    u += 0x7FFFu + ((u >> 16) & 1u);   // round-to-nearest-even
    return (unsigned short)(u >> 16);
}

union FragBF { v16bf v; uint4 q[2]; };

// A/B fragment gather from LDS for v_wmma_f32_16x16x32_bf16.
// Per ISA 16-bit A layout: lanes 0-15 row=lane hold K {0..7, 16..23},
// lanes 16-31 row=lane-16 hold K {8..15, 24..31} -> two b128 LDS reads.
__device__ __forceinline__ v16bf lds_frag(const unsigned short* base, int row0,
                                          int ld, int k0) {
    const int lane = threadIdx.x & 31;
    const int half = lane >> 4;
    const int r    = lane & 15;
    const unsigned short* p = base + (row0 + r) * ld + k0;
    FragBF f;
    f.q[0] = *(const uint4*)(p + half * 8);
    f.q[1] = *(const uint4*)(p + 16 + half * 8);
    return f.v;
}

#define WMMA_BF16(A, B, C) \
    __builtin_amdgcn_wmma_f32_16x16x32_bf16(false, (A), false, (B), (short)0, (C), false, false)

// ---------------------------------------------------------------------------
// 1) x[B,E,S] + pos[S,E]  ->  Xb bf16 [M=25600, 1024]
// ---------------------------------------------------------------------------
__global__ __launch_bounds__(256) void convert_x_kernel(
        const float* __restrict__ x, const float* __restrict__ pos,
        unsigned short* __restrict__ Xb) {
    const int m = blockIdx.x;                       // b*100 + s
    const int e = blockIdx.y * 256 + threadIdx.x;
    const int b = m / SEQ;
    const int s = m - b * SEQ;
    float v = x[((size_t)b * EMBED + e) * SEQ + s] + pos[(size_t)s * EMBED + e];
    Xb[(size_t)m * EMBED + e] = f32_to_bf16(v);
}

// ---------------------------------------------------------------------------
// 2) W[K,N] f32 -> Wt bf16 [N,K]  (transposed so B-fragments are contiguous-K)
// ---------------------------------------------------------------------------
__global__ __launch_bounds__(256) void convert_w_kernel(
        const float* __restrict__ W, unsigned short* __restrict__ Wt) {
    const int n = blockIdx.x;
    const int k = blockIdx.y * 256 + threadIdx.x;
    Wt[(size_t)n * EMBED + k] = f32_to_bf16(W[(size_t)k * EMBED + n]);
}

// ---------------------------------------------------------------------------
// Tile staging: each of the 256 threads moves two A uint4 chunks and one
// B uint4 chunk (A tile 128x32 bf16, B tile 64x32 bf16, LDS stride 40).
// ---------------------------------------------------------------------------
__device__ __forceinline__ void stage_tiles(
        const unsigned short* __restrict__ A,
        const unsigned short* __restrict__ Bt,
        int m0, int n0, int k0,
        unsigned short* sA, unsigned short* sB) {
    const int tid = threadIdx.x;
    const int row = tid >> 2;          // 0..63
    const int q   = (tid & 3) * 8;     // bf16 offset of the 16B chunk
#if GEMM_ASYNC
    __builtin_amdgcn_global_load_async_to_lds_b128(
        gcast(A + (size_t)(m0 + row) * EMBED + k0 + q),
        lcast(sA + row * LDA + q), 0, 0);
    __builtin_amdgcn_global_load_async_to_lds_b128(
        gcast(A + (size_t)(m0 + row + 64) * EMBED + k0 + q),
        lcast(sA + (row + 64) * LDA + q), 0, 0);
    __builtin_amdgcn_global_load_async_to_lds_b128(
        gcast(Bt + (size_t)(n0 + row) * EMBED + k0 + q),
        lcast(sB + row * LDA + q), 0, 0);
#else
    const uint4 a0 = *(const uint4*)(A + (size_t)(m0 + row) * EMBED + k0 + q);
    const uint4 a1 = *(const uint4*)(A + (size_t)(m0 + row + 64) * EMBED + k0 + q);
    const uint4 b0 = *(const uint4*)(Bt + (size_t)(n0 + row) * EMBED + k0 + q);
    *(uint4*)(sA + row * LDA + q)        = a0;
    *(uint4*)(sA + (row + 64) * LDA + q) = a1;
    *(uint4*)(sB + row * LDA + q)        = b0;
#endif
}

__device__ __forceinline__ void stage_wait() {
#if GEMM_ASYNC
    __builtin_amdgcn_s_wait_asynccnt(0);
#endif
}

// ---------------------------------------------------------------------------
// Shared WMMA GEMM mainloop: C[128x64] block, 8 waves, 32x32 per wave,
// double-buffered LDS (stage slice i+1 while WMMA consumes slice i).
// ---------------------------------------------------------------------------
__device__ __forceinline__ void gemm_mainloop(
        const unsigned short* __restrict__ A,   // [M,1024] bf16 row-major
        const unsigned short* __restrict__ Bt,  // [N,1024] bf16 (W transposed)
        int m0, int n0, v8f c[2][2],
        unsigned short* sA, unsigned short* sB) {
    const int wave = threadIdx.x >> 5;
    const int wm   = (wave & 3) * 32;
    const int wn   = (wave >> 2) * 32;

    stage_tiles(A, Bt, m0, n0, 0, sA, sB);
    stage_wait();
    __syncthreads();

    for (int it = 0; it < EMBED / 32; ++it) {
        const int cur = it & 1;
        unsigned short* cA = sA + cur * TILE_A;
        unsigned short* cB = sB + cur * TILE_B;
        if (it + 1 < EMBED / 32)
            stage_tiles(A, Bt, m0, n0, (it + 1) * 32,
                        sA + (cur ^ 1) * TILE_A, sB + (cur ^ 1) * TILE_B);

        v16bf a0 = lds_frag(cA, wm,      LDA, 0);
        v16bf a1 = lds_frag(cA, wm + 16, LDA, 0);
        v16bf b0 = lds_frag(cB, wn,      LDA, 0);
        v16bf b1 = lds_frag(cB, wn + 16, LDA, 0);
        c[0][0] = WMMA_BF16(a0, b0, c[0][0]);
        c[0][1] = WMMA_BF16(a0, b1, c[0][1]);
        c[1][0] = WMMA_BF16(a1, b0, c[1][0]);
        c[1][1] = WMMA_BF16(a1, b1, c[1][1]);

        stage_wait();          // my async writes to the other buffer landed
        __syncthreads();       // everyone done reading cur + next tile ready
    }
}

// ---------------------------------------------------------------------------
// 3) Q/K/V projection: Out bf16 [M,1024] = Xb @ W + bias
// ---------------------------------------------------------------------------
__global__ __launch_bounds__(256) void gemm_qkv_kernel(
        const unsigned short* __restrict__ A, const unsigned short* __restrict__ Bt,
        const float* __restrict__ bias, unsigned short* __restrict__ Out) {
    __shared__ unsigned short sA[2 * TILE_A];
    __shared__ unsigned short sB[2 * TILE_B];
    const v8f vz = {0.f, 0.f, 0.f, 0.f, 0.f, 0.f, 0.f, 0.f};
    v8f c[2][2] = {{vz, vz}, {vz, vz}};
    const int m0 = blockIdx.x * 128, n0 = blockIdx.y * 64;
    gemm_mainloop(A, Bt, m0, n0, c, sA, sB);

    const int tid = threadIdx.x, wave = tid >> 5, lane = tid & 31;
    const int wm = (wave & 3) * 32, wn = (wave >> 2) * 32;
    const int nl = lane & 15, mh = (lane >> 4) * 8;
    for (int tm = 0; tm < 2; tm++)
        for (int tn = 0; tn < 2; tn++)
            for (int r = 0; r < 8; r++) {
                const int m = m0 + wm + tm * 16 + mh + r;
                const int n = n0 + wn + tn * 16 + nl;
                Out[(size_t)m * EMBED + n] = f32_to_bf16(c[tm][tn][r] + bias[n]);
            }
}

// ---------------------------------------------------------------------------
// 5) Output projection with transposed store: out[b,e,s] = Att @ Wo + bo
// ---------------------------------------------------------------------------
__global__ __launch_bounds__(256) void gemm_out_kernel(
        const unsigned short* __restrict__ A, const unsigned short* __restrict__ Bt,
        const float* __restrict__ bias, float* __restrict__ Out) {
    __shared__ unsigned short sA[2 * TILE_A];
    __shared__ unsigned short sB[2 * TILE_B];
    const v8f vz = {0.f, 0.f, 0.f, 0.f, 0.f, 0.f, 0.f, 0.f};
    v8f c[2][2] = {{vz, vz}, {vz, vz}};
    const int m0 = blockIdx.x * 128, n0 = blockIdx.y * 64;
    gemm_mainloop(A, Bt, m0, n0, c, sA, sB);

    const int tid = threadIdx.x, wave = tid >> 5, lane = tid & 31;
    const int wm = (wave & 3) * 32, wn = (wave >> 2) * 32;
    const int nl = lane & 15, mh = (lane >> 4) * 8;
    for (int tm = 0; tm < 2; tm++)
        for (int tn = 0; tn < 2; tn++)
            for (int r = 0; r < 8; r++) {
                const int m = m0 + wm + tm * 16 + mh + r;
                const int n = n0 + wn + tn * 16 + nl;
                const int b = m / SEQ, s = m - b * SEQ;
                Out[(size_t)b * (EMBED * SEQ) + (size_t)n * SEQ + s] =
                    c[tm][tn][r] + bias[n];
            }
}

// ---------------------------------------------------------------------------
// 4) Fused attention per (b,h): QK^T -> softmax -> P@V, all WMMA.
//    LDS plan (dynamic, 101632 B):
//      phase1: sQ[112x72]@0  sK[112x72]@16128  sVt[64x136]@32256  sS f32[112x116]@49664
//      phase2: sP[112x136]@0 (reuses dead Q/K region)
// ---------------------------------------------------------------------------
__global__ __launch_bounds__(128) void attention_kernel(
        const unsigned short* __restrict__ Qb, const unsigned short* __restrict__ Kb,
        const unsigned short* __restrict__ Vb, float* __restrict__ attn,
        unsigned short* __restrict__ Attb) {
    extern __shared__ unsigned char smem[];
    unsigned short* sQ  = (unsigned short*)smem;
    unsigned short* sK  = (unsigned short*)(smem + 16128);
    unsigned short* sVt = (unsigned short*)(smem + 32256);
    float*          sS  = (float*)(smem + 49664);
    unsigned short* sP  = (unsigned short*)smem;

    const int tid  = threadIdx.x;
    const int wave = tid >> 5;
    const int lane = tid & 31;
    const int b = blockIdx.x >> 4;
    const int h = blockIdx.x & 15;
    const size_t base = (size_t)b * SEQ * EMBED + h * HDIM;

    // Stage Q, K (zero-pad rows 100..111)
    for (int t = tid; t < SP * 8; t += 128) {
        const int s = t >> 3, dq = (t & 7) * 8;
        uint4 q4 = {0, 0, 0, 0}, k4 = {0, 0, 0, 0};
        if (s < SEQ) {
            q4 = *(const uint4*)(Qb + base + (size_t)s * EMBED + dq);
            k4 = *(const uint4*)(Kb + base + (size_t)s * EMBED + dq);
        }
        *(uint4*)(sQ + s * 72 + dq) = q4;
        *(uint4*)(sK + s * 72 + dq) = k4;
    }
    // Stage V transposed: sVt[d][s'], zero-pad s' in [100,128)
    for (int t = tid; t < HDIM * KP; t += 128) {
        const int d = t >> 7, s = t & 127;
        unsigned short v = 0;
        if (s < SEQ) v = Vb[base + (size_t)s * EMBED + d];
        sVt[d * 136 + s] = v;
    }
    __syncthreads();

    // scores = Q K^T / 8  (7x7 tile grid, k=64 -> 2 wmma per tile)
    const int nl = lane & 15, mh = (lane >> 4) * 8;
    const v8f vz = {0.f, 0.f, 0.f, 0.f, 0.f, 0.f, 0.f, 0.f};
    for (int tr = wave; tr < 7; tr += 4) {
        for (int tc = 0; tc < 7; tc++) {
            v8f acc = vz;
            for (int ks = 0; ks < 2; ks++) {
                v16bf a  = lds_frag(sQ, tr * 16, 72, ks * 32);
                v16bf bb = lds_frag(sK, tc * 16, 72, ks * 32);
                acc = WMMA_BF16(a, bb, acc);
            }
            for (int r = 0; r < 8; r++)
                sS[(tr * 16 + mh + r) * 116 + tc * 16 + nl] = acc[r] * 0.125f;
        }
    }
    __syncthreads();

    // Row softmax over the 100 real columns; emit f32 attn + bf16 P
    if (tid < SEQ) {
        float* row = sS + tid * 116;
        float mx = -3.0e38f;
        for (int c2 = 0; c2 < SEQ; c2++) mx = fmaxf(mx, row[c2]);
        float sum = 0.f;
        for (int c2 = 0; c2 < SEQ; c2++) {
            const float e = __expf(row[c2] - mx);
            row[c2] = e;
            sum += e;
        }
        const float inv = 1.0f / sum;
        float* arow = attn + (((size_t)(b * HEADS + h)) * SEQ + tid) * SEQ;
        for (int c2 = 0; c2 < SEQ; c2++) {
            const float p = row[c2] * inv;
            arow[c2] = p;
            sP[tid * 136 + c2] = f32_to_bf16(p);
        }
        for (int c2 = SEQ; c2 < KP; c2++) sP[tid * 136 + c2] = 0;
    } else if (tid < SP) {
        for (int c2 = 0; c2 < KP; c2++) sP[tid * 136 + c2] = 0;
    }
    __syncthreads();

    // attended = P @ V  (7 s-tiles x 4 d-tiles, one d-tile per wave, k=128)
    const int td = wave;
    for (int ts = 0; ts < 7; ts++) {
        v8f acc = vz;
        for (int ks = 0; ks < 4; ks++) {
            v16bf a  = lds_frag(sP, ts * 16, 136, ks * 32);
            v16bf bb = lds_frag(sVt, td * 16, 136, ks * 32);
            acc = WMMA_BF16(a, bb, acc);
        }
        for (int r = 0; r < 8; r++) {
            const int s = ts * 16 + mh + r;
            if (s < SEQ)
                Attb[base + (size_t)s * EMBED + td * 16 + nl] =
                    f32_to_bf16(acc[r]);
        }
    }
}

// ---------------------------------------------------------------------------
// Host driver
// ---------------------------------------------------------------------------
extern "C" void kernel_launch(void* const* d_in, const int* in_sizes, int n_in,
                              void* d_out, int out_size, void* d_ws, size_t ws_size,
                              hipStream_t stream) {
    (void)in_sizes; (void)n_in; (void)out_size; (void)ws_size;
    const float* x   = (const float*)d_in[0];
    const float* Wq  = (const float*)d_in[1];
    const float* bq  = (const float*)d_in[2];
    const float* Wk  = (const float*)d_in[3];
    const float* bk  = (const float*)d_in[4];
    const float* Wv  = (const float*)d_in[5];
    const float* bv  = (const float*)d_in[6];
    const float* Wo  = (const float*)d_in[7];
    const float* bo  = (const float*)d_in[8];
    const float* pos = (const float*)d_in[9];

    // Workspace layout (bf16 buffers), ~258 MB total
    unsigned char* ws = (unsigned char*)d_ws;
    const size_t XB_BYTES = (size_t)M_TOT * EMBED * 2;  // 52,428,800
    const size_t W_BYTES  = (size_t)EMBED * EMBED * 2;  //  2,097,152
    unsigned short* Xb   = (unsigned short*)(ws);
    unsigned short* Wqt  = (unsigned short*)(ws + XB_BYTES);
    unsigned short* Wkt  = (unsigned short*)(ws + XB_BYTES + 1 * W_BYTES);
    unsigned short* Wvt  = (unsigned short*)(ws + XB_BYTES + 2 * W_BYTES);
    unsigned short* Wot  = (unsigned short*)(ws + XB_BYTES + 3 * W_BYTES);
    unsigned short* Qbuf = (unsigned short*)(ws + XB_BYTES + 4 * W_BYTES);
    unsigned short* Kbuf = (unsigned short*)(ws + 2 * XB_BYTES + 4 * W_BYTES);
    unsigned short* Vbuf = (unsigned short*)(ws + 3 * XB_BYTES + 4 * W_BYTES);
    unsigned short* Attb = (unsigned short*)(ws + 4 * XB_BYTES + 4 * W_BYTES);

    float* out  = (float*)d_out;                               // [B,E,H,W]
    float* attn = out + (size_t)BATCH * EMBED * SEQ;           // [B,H,S,S]

    convert_x_kernel<<<dim3(M_TOT, EMBED / 256), 256, 0, stream>>>(x, pos, Xb);
    convert_w_kernel<<<dim3(EMBED, EMBED / 256), 256, 0, stream>>>(Wq, Wqt);
    convert_w_kernel<<<dim3(EMBED, EMBED / 256), 256, 0, stream>>>(Wk, Wkt);
    convert_w_kernel<<<dim3(EMBED, EMBED / 256), 256, 0, stream>>>(Wv, Wvt);
    convert_w_kernel<<<dim3(EMBED, EMBED / 256), 256, 0, stream>>>(Wo, Wot);

    const dim3 gg(M_TOT / 128, EMBED / 64);   // 200 x 16 blocks
    gemm_qkv_kernel<<<gg, 256, 0, stream>>>(Xb, Wqt, bq, Qbuf);
    gemm_qkv_kernel<<<gg, 256, 0, stream>>>(Xb, Wkt, bk, Kbuf);
    gemm_qkv_kernel<<<gg, 256, 0, stream>>>(Xb, Wvt, bv, Vbuf);

    attention_kernel<<<dim3(BATCH * HEADS), 128, 101632, stream>>>(
        Qbuf, Kbuf, Vbuf, attn, Attb);

    gemm_out_kernel<<<gg, 256, 0, stream>>>(Attb, Wot, bo, out);
}
